// MultiQueryAttention_13632226197804
// MI455X (gfx1250) — matrix operand order repaired and checked by reading, not compiled, not run on hardware
//
#include <hip/hip_runtime.h>
#include <hip/hip_bf16.h>

typedef __bf16 bf16;
typedef __attribute__((ext_vector_type(8)))  __bf16 v8bf;
typedef __attribute__((ext_vector_type(16))) __bf16 v16bf;
typedef __attribute__((ext_vector_type(8)))  float  v8f;

constexpr int cB = 8, cS = 2048, cM = 256, cD = 512, cP = 512, cH = 8;
constexpr float cScale = 0.04419417382415922f; // 1/sqrt(512)

// ---------------------------------------------------------------------------
// A-fragment load (contiguous K order, matching B): lane holds row m=lane&15,
// K = {kb..kb+15}, kb=(lane<16?0:16)  -> two contiguous 16B loads.
// ---------------------------------------------------------------------------
__device__ __forceinline__ v16bf load_afrag(const bf16* __restrict__ p) {
  v8bf a0 = *(const v8bf*)p;
  v8bf a1 = *(const v8bf*)(p + 8);
  return __builtin_shufflevector(a0, a1, 0,1,2,3,4,5,6,7,8,9,10,11,12,13,14,15);
}

// ---------------------------------------------------------------------------
// Core: one wave computes a 64x64 macro-tile (4x4 grid of 16x16 WMMA tiles).
// B-frags (4) are preloaded; A-frags are streamed one at a time to keep the
// peak VGPR live set ~175 (16 f32 accs = 128 VGPRs dominate).
// Per K-step of 32: 8KB of loads -> 16 WMMAs = 32 FLOP/byte.
// ---------------------------------------------------------------------------
__device__ __forceinline__ void wmma_block44(const bf16* __restrict__ A, long long lda,
                                             const bf16* __restrict__ BT, long long ldb,
                                             int K, int mt, int nt,
                                             int lm, int half, v8f acc[16]) {
  const bf16* ar0 = A  + (long long)(mt + lm) * lda;
  const bf16* br0 = BT + (long long)(nt + lm) * ldb;
  const int ka = half * 16, kb = half * 16;
  const long long la16 = 16 * lda, lb16 = 16 * ldb;
#pragma unroll 1
  for (int k = 0; k < K; k += 32) {
    v16bf bv[4];
#pragma unroll
    for (int j = 0; j < 4; ++j)
      bv[j] = *(const v16bf*)(br0 + j * lb16 + k + kb);
#pragma unroll
    for (int i = 0; i < 4; ++i) {
      v16bf av = load_afrag(ar0 + i * la16 + k + ka);
#pragma unroll
      for (int j = 0; j < 4; ++j)
        acc[i * 4 + j] = __builtin_amdgcn_wmma_f32_16x16x32_bf16(
            false, av, false, bv[j], (short)0, acc[i * 4 + j], false, false);
    }
  }
}

// Map wave -> 64x64 macro-tile; false if out of range (wave-uniform).
__device__ __forceinline__ bool tile_setup(int Mr, int N, int& mt, int& nt,
                                           int& lm, int& half) {
  int wave = threadIdx.x >> 5;
  int lane = threadIdx.x & 31;
  int tile = blockIdx.x * (blockDim.x >> 5) + wave;
  int tn = N >> 6, tm = Mr >> 6;
  if (tile >= tm * tn) return false;
  mt = (tile / tn) << 6; nt = (tile % tn) << 6;
  lm = lane & 15; half = lane >> 4;
  return true;
}

// ---------------------------------------------------------------------------
// Generic GEMM, bf16 output, row-major store, optional f32 bias[n].
// ---------------------------------------------------------------------------
__global__ void __launch_bounds__(128, 1)
k_gemm_bf16(const bf16* __restrict__ A, long long lda,
            long long sAy, long long sAz,
            const bf16* __restrict__ BT, long long ldb,
            long long sBy, long long sBz,
            const float* __restrict__ bias, long long sBiasZ,
            bf16* __restrict__ C, long long ldc,
            long long sCy, long long sCz,
            int Mr, int N, int K) {
  int mt, nt, lm, half;
  if (!tile_setup(Mr, N, mt, nt, lm, half)) return;
  A  += blockIdx.y * sAy + blockIdx.z * sAz;
  BT += blockIdx.y * sBy + blockIdx.z * sBz;
  C  += blockIdx.y * sCy + blockIdx.z * sCz;
  v8f acc[16] = {};
  wmma_block44(A, lda, BT, ldb, K, mt, nt, lm, half, acc);
#pragma unroll
  for (int i = 0; i < 4; ++i) {
    int mb = mt + i * 16 + half * 8;
#pragma unroll
    for (int j = 0; j < 4; ++j) {
      int n = nt + j * 16 + lm;
      float bia = bias ? bias[blockIdx.z * sBiasZ + n] : 0.f;
#pragma unroll
      for (int r = 0; r < 8; ++r)
        C[(long long)(mb + r) * ldc + n] = (bf16)(acc[i * 4 + j][r] + bia);
    }
  }
}

// Same, transposed bf16 store: CT[n][m] (contiguous 16B per lane per tile).
__global__ void __launch_bounds__(128, 1)
k_gemm_bf16_trans(const bf16* __restrict__ A, long long lda,
                  long long sAy,
                  const bf16* __restrict__ BT, long long ldb,
                  const float* __restrict__ bias,
                  bf16* __restrict__ CT, long long ldct,
                  long long sCy, int Mr, int N, int K) {
  int mt, nt, lm, half;
  if (!tile_setup(Mr, N, mt, nt, lm, half)) return;
  A  += blockIdx.y * sAy;
  CT += blockIdx.y * sCy;
  v8f acc[16] = {};
  wmma_block44(A, lda, BT, ldb, K, mt, nt, lm, half, acc);
#pragma unroll
  for (int j = 0; j < 4; ++j) {
    int n = nt + j * 16 + lm;
    float bia = bias ? bias[n] : 0.f;
#pragma unroll
    for (int i = 0; i < 4; ++i) {
      int mb = mt + i * 16 + half * 8;
      v8bf o;
#pragma unroll
      for (int r = 0; r < 8; ++r) o[r] = (bf16)(acc[i * 4 + j][r] + bia);
      *(v8bf*)(CT + (long long)n * ldct + mb) = o;
    }
  }
}

// Generic GEMM, f32 output, optional bias (scores & final projection).
__global__ void __launch_bounds__(128, 1)
k_gemm_f32(const bf16* __restrict__ A, long long lda,
           long long sAy, long long sAz,
           const bf16* __restrict__ BT, long long ldb,
           long long sBy, long long sBz,
           const float* __restrict__ bias,
           float* __restrict__ C, long long ldc,
           long long sCy, long long sCz,
           int Mr, int N, int K) {
  int mt, nt, lm, half;
  if (!tile_setup(Mr, N, mt, nt, lm, half)) return;
  A  += blockIdx.y * sAy + blockIdx.z * sAz;
  BT += blockIdx.y * sBy + blockIdx.z * sBz;
  C  += blockIdx.y * sCy + blockIdx.z * sCz;
  v8f acc[16] = {};
  wmma_block44(A, lda, BT, ldb, K, mt, nt, lm, half, acc);
#pragma unroll
  for (int i = 0; i < 4; ++i) {
    int mb = mt + i * 16 + half * 8;
#pragma unroll
    for (int j = 0; j < 4; ++j) {
      int n = nt + j * 16 + lm;
      float bia = bias ? bias[n] : 0.f;
#pragma unroll
      for (int r = 0; r < 8; ++r)
        C[(long long)(mb + r) * ldc + n] = acc[i * 4 + j][r] + bia;
    }
  }
}

// Value GEMM: val = attn(M,S) @ ivT(P,S)^T + self_attn*mv, bf16 into
// concat (B, M, H*P) at column offset h*P.
__global__ void __launch_bounds__(128, 1)
k_gemm_val(const bf16* __restrict__ attn,
           const bf16* __restrict__ ivT,
           const bf16* __restrict__ mv,
           const float* __restrict__ selfat,
           bf16* __restrict__ concat) {
  int b = blockIdx.y, h = blockIdx.z;
  int mt, nt, lm, half;
  if (!tile_setup(cM, cP, mt, nt, lm, half)) return;
  const bf16* A  = attn + (long long)(b * cH + h) * cM * cS;
  const bf16* BT = ivT  + (long long)b * cP * cS;
  v8f acc[16] = {};
  wmma_block44(A, cS, BT, cS, cS, mt, nt, lm, half, acc);
  const float* sa = selfat + (long long)(b * cH + h) * cM;
#pragma unroll
  for (int i = 0; i < 4; ++i) {
    int mb = mt + i * 16 + half * 8;
#pragma unroll
    for (int j = 0; j < 4; ++j) {
      int n = nt + j * 16 + lm;
#pragma unroll
      for (int r = 0; r < 8; ++r) {
        int m = mb + r;
        float v = acc[i * 4 + j][r] +
                  sa[m] * (float)mv[((long long)b * cM + m) * cP + n];
        concat[((long long)b * cM + m) * (cH * cP) + (long long)h * cP + n] = (bf16)v;
      }
    }
  }
}

// Softmax over S+1 per (b,h,m) row; fuses self-score dot(q,mk).
__global__ void k_softmax(const float* __restrict__ scores,
                          const bf16* __restrict__ q,
                          const bf16* __restrict__ mk,
                          bf16* __restrict__ attn,
                          float* __restrict__ selfat) {
  __shared__ float red[256];
  __shared__ float rowe[cS];
  int row = blockIdx.x;              // b*H*M + h*M + m
  int b = row / (cH * cM);
  int m = row % cM;
  int t = threadIdx.x;
  const bf16* qr  = q  + (long long)row * cP;
  const bf16* mkr = mk + ((long long)b * cM + m) * cP;
  float part = 0.f;
  for (int p = t; p < cP; p += 256) part += (float)qr[p] * (float)mkr[p];
  red[t] = part; __syncthreads();
  for (int s2 = 128; s2 > 0; s2 >>= 1) {
    if (t < s2) red[t] += red[t + s2];
    __syncthreads();
  }
  float ss = red[0] * cScale;        // scaled self score
  __syncthreads();
  const float* sr = scores + (long long)row * cS;
  float lmax = ss;
  for (int s = t; s < cS; s += 256) lmax = fmaxf(lmax, sr[s] * cScale);
  red[t] = lmax; __syncthreads();
  for (int s2 = 128; s2 > 0; s2 >>= 1) {
    if (t < s2) red[t] = fmaxf(red[t], red[t + s2]);
    __syncthreads();
  }
  float gmax = red[0]; __syncthreads();
  float lsum = 0.f;
  for (int s = t; s < cS; s += 256) {
    float e = __expf(sr[s] * cScale - gmax);
    rowe[s] = e; lsum += e;
  }
  if (t == 0) lsum += __expf(ss - gmax);
  red[t] = lsum; __syncthreads();
  for (int s2 = 128; s2 > 0; s2 >>= 1) {
    if (t < s2) red[t] += red[t + s2];
    __syncthreads();
  }
  float inv = 1.f / red[0];
  bf16* ar = attn + (long long)row * cS;
  for (int s = t; s < cS; s += 256) ar[s] = (bf16)(rowe[s] * inv);
  if (t == 0) selfat[row] = __expf(ss - gmax) * inv;
}

// Elementwise cast f32 -> bf16.
__global__ void k_cast_bf16(const float* __restrict__ in, bf16* __restrict__ out,
                            long long n) {
  long long i = (long long)blockIdx.x * blockDim.x + threadIdx.x;
  if (i < n) out[i] = (bf16)in[i];
}

// Transpose + cast: in (R,C) f32 -> out (C,R) bf16, batched via blockIdx.y.
__global__ void k_transpose_cast(const float* __restrict__ in,
                                 bf16* __restrict__ out, int R, int C) {
  long long base = (long long)blockIdx.y * R * C;
  long long i = (long long)blockIdx.x * blockDim.x + threadIdx.x;
  if (i >= (long long)R * C) return;
  int r = (int)(i / C), c = (int)(i % C);
  out[base + (long long)c * R + r] = (bf16)in[base + i];
}

extern "C" void kernel_launch(void* const* d_in, const int* in_sizes, int n_in,
                              void* d_out, int out_size, void* d_ws, size_t ws_size,
                              hipStream_t stream) {
  (void)in_sizes; (void)n_in; (void)out_size; (void)ws_size;
  const float* x   = (const float*)d_in[0];   // (B,S,D)
  const float* mem = (const float*)d_in[1];   // (B,M,D)
  const float* Wk  = (const float*)d_in[2];   // (D,P)
  const float* bk  = (const float*)d_in[3];
  const float* Wv  = (const float*)d_in[4];
  const float* bv  = (const float*)d_in[5];
  const float* Wq  = (const float*)d_in[6];   // (H,D,P)
  const float* bq  = (const float*)d_in[7];   // (H,P)
  const float* Wo  = (const float*)d_in[8];   // (H*P,P)
  const float* bo  = (const float*)d_in[9];

  char* p = (char*)d_ws;
  auto alloc = [&](long long bytes) -> void* {
    void* r = (void*)p;
    p += (bytes + 255) & ~255LL;
    return r;
  };
  const long long nX   = (long long)cB * cS * cD;
  const long long nMem = (long long)cB * cM * cD;
  const long long nW   = (long long)cD * cP;
  const long long nWq  = (long long)cH * cD * cP;
  const long long nWo  = (long long)cH * cP * cP;
  const long long nIK  = (long long)cB * cS * cP;
  const long long nMK  = (long long)cB * cM * cP;
  const long long nQ   = (long long)cB * cH * cM * cP;
  const long long nSc  = (long long)cB * cH * cM * cS;
  const long long nCat = (long long)cB * cM * cH * cP;

  bf16* xb     = (bf16*)alloc(nX * 2);
  bf16* memb   = (bf16*)alloc(nMem * 2);
  bf16* WkT    = (bf16*)alloc(nW * 2);
  bf16* WvT    = (bf16*)alloc(nW * 2);
  bf16* WqT    = (bf16*)alloc(nWq * 2);
  bf16* WoT    = (bf16*)alloc(nWo * 2);
  bf16* ikb    = (bf16*)alloc(nIK * 2);
  bf16* mkb    = (bf16*)alloc(nMK * 2);
  bf16* ivTb   = (bf16*)alloc(nIK * 2);   // (B,P,S)
  bf16* mvb    = (bf16*)alloc(nMK * 2);
  bf16* qb     = (bf16*)alloc(nQ * 2);
  float* scores = (float*)alloc(nSc * 4);
  bf16* attnb  = (bf16*)alloc(nSc * 2);
  float* selfat = (float*)alloc((long long)cB * cH * cM * 4);
  bf16* catb   = (bf16*)alloc(nCat * 2);

  // 1) casts & weight transposes
  k_cast_bf16<<<dim3((unsigned)((nX + 255) / 256)), 256, 0, stream>>>(x, xb, nX);
  k_cast_bf16<<<dim3((unsigned)((nMem + 255) / 256)), 256, 0, stream>>>(mem, memb, nMem);
  k_transpose_cast<<<dim3((unsigned)((nW + 255) / 256), 1), 256, 0, stream>>>(Wk, WkT, cD, cP);
  k_transpose_cast<<<dim3((unsigned)((nW + 255) / 256), 1), 256, 0, stream>>>(Wv, WvT, cD, cP);
  k_transpose_cast<<<dim3((unsigned)((nW + 255) / 256), cH), 256, 0, stream>>>(Wq, WqT, cD, cP);
  k_transpose_cast<<<dim3((unsigned)((nWo + 255) / 256), 1), 256, 0, stream>>>(Wo, WoT, cH * cP, cP);

  // 2) projections (64x64 macro-tiles, 4 waves/block)
  // ik = x @ Wk + bk : (B*S, P)  tiles = 256*8 -> 512 blocks
  k_gemm_bf16<<<dim3(512), 128, 0, stream>>>(
      xb, cD, 0, 0, WkT, cD, 0, 0, bk, 0,
      ikb, cP, 0, 0, cB * cS, cP, cD);
  // mk = mem @ Wk + bk : (B*M, P)  tiles = 32*8 -> 64 blocks
  k_gemm_bf16<<<dim3(64), 128, 0, stream>>>(
      memb, cD, 0, 0, WkT, cD, 0, 0, bk, 0,
      mkb, cP, 0, 0, cB * cM, cP, cD);
  // mv = mem @ Wv + bv : (B*M, P)
  k_gemm_bf16<<<dim3(64), 128, 0, stream>>>(
      memb, cD, 0, 0, WvT, cD, 0, 0, bv, 0,
      mvb, cP, 0, 0, cB * cM, cP, cD);
  // ivT[b] = (x[b] @ Wv + bv)^T : (P,S) per batch, tiles = 32*8 -> 64 blocks
  k_gemm_bf16_trans<<<dim3(64, cB), 128, 0, stream>>>(
      xb, cD, (long long)cS * cD, WvT, cD, bv,
      ivTb, cS, (long long)cP * cS, cS, cP, cD);
  // q[b,h] = mem[b] @ Wq[h] + bq[h] : (M,P)  tiles = 4*8 -> 8 blocks
  k_gemm_bf16<<<dim3(8, cB, cH), 128, 0, stream>>>(
      memb, cD, (long long)cM * cD, 0,
      WqT, cD, 0, (long long)cP * cD,
      bq, cP,
      qb, cP, (long long)cH * cM * cP, (long long)cM * cP,
      cM, cP, cD);

  // 3) scores[b,h] = q[b,h] @ ik[b]^T : (M,S) f32  tiles = 4*32 -> 32 blocks
  k_gemm_f32<<<dim3(32, cB, cH), 128, 0, stream>>>(
      qb, cP, (long long)cH * cM * cP, (long long)cM * cP,
      ikb, cP, (long long)cS * cP, 0,
      nullptr,
      scores, cS, (long long)cH * cM * cS, (long long)cM * cS,
      cM, cS, cP);

  // 4) softmax over S+1 (fused self-score)
  k_softmax<<<dim3(cB * cH * cM), 256, 0, stream>>>(scores, qb, mkb, attnb, selfat);

  // 5) val = attn @ iv + self_attn*mv -> concat (B, M, H*P)  tiles 4*8 -> 8 blocks
  k_gemm_val<<<dim3(8, cB, cH), 128, 0, stream>>>(attnb, ivTb, mvb, selfat, catb);

  // 6) out = concat @ Wo + bo : (B*M, P) f32  tiles = 32*8 -> 64 blocks
  k_gemm_f32<<<dim3(64), 128, 0, stream>>>(
      catb, cH * cP, 0, 0, WoT, cH * cP, 0, 0, bo,
      (float*)d_out, cP, 0, 0, cB * cM, cP, cH * cP);
}
